// MultiheadAttention_73014444032345
// MI455X (gfx1250) — compile-verified, matrix operand order checked by reading, not run on hardware
//
#include <hip/hip_runtime.h>
#include <hip/hip_bf16.h>
#include <stdint.h>

typedef __bf16 bf16;
typedef bf16  v16bf __attribute__((ext_vector_type(16)));
typedef bf16  bf8v  __attribute__((ext_vector_type(8)));
typedef float v8f   __attribute__((ext_vector_type(8)));
typedef unsigned int u32x4 __attribute__((ext_vector_type(4)));
typedef int   i32x4 __attribute__((ext_vector_type(4)));
typedef int   i32x8 __attribute__((ext_vector_type(8)));

// ---- constants for this problem ----
#define BQ   4
#define SQ   2048
#define DIN  1024
#define DEMB 1024
#define NH   16
#define HD   64
#define NTOK (BQ * SQ)          // 8192
#define QKVC (3 * DEMB)         // 3072

#if defined(__gfx1250__) && __has_builtin(__builtin_amdgcn_tensor_load_to_lds)
#define HAVE_TDM 1
#else
#define HAVE_TDM 0
#endif

// combine two 16B LDS chunks into one 16x bf16 WMMA fragment
static __device__ __forceinline__ v16bf load16(const bf16* lo, const bf16* hi) {
    bf8v a = *(const bf8v*)lo;
    bf8v b = *(const bf8v*)hi;
    return __builtin_shufflevector(a, b, 0,1,2,3,4,5,6,7,8,9,10,11,12,13,14,15);
}

// ---- async global -> LDS (16B per lane), tracked by ASYNCcnt ----
static __device__ __forceinline__ void async_load_b128(const bf16* gptr, bf16* lptr) {
    unsigned lds = (unsigned)(uintptr_t)lptr;   // LDS aperture: addr[31:0] is LDS offset
    asm volatile("global_load_async_to_lds_b128 %0, %1, off"
                 :: "v"(lds), "v"(gptr) : "memory");
}
static __device__ __forceinline__ void wait_async() {
    asm volatile("s_wait_asynccnt 0x0" ::: "memory");
}

#if HAVE_TDM
// ---- Tensor Data Mover: 2-D bf16 tile global -> LDS (one DMA per call) ----
// pad_int_enc: LDS pad every (2<<enc) DWORDs; pad_amt_enc: skip (enc+1) DWORDs.
static __device__ __forceinline__ void tdm_load_2d_bf16(
    unsigned lds_addr, const bf16* gptr,
    unsigned tensor_d0, unsigned tensor_d1,
    unsigned tile_d0, unsigned tile_d1,
    unsigned d0_stride,
    bool pad_en, unsigned pad_int_enc, unsigned pad_amt_enc)
{
    unsigned long long ga = (unsigned long long)(uintptr_t)gptr;
    u32x4 g0;
    g0.x = 1u;                                            // count=1, user mode
    g0.y = lds_addr;                                      // lds_addr [63:32]
    g0.z = (unsigned)ga;                                  // global_addr[31:0]
    g0.w = (unsigned)((ga >> 32) & 0x01FFFFFFu) | (2u << 30);  // addr[56:32], type=2
    i32x8 g1;
    unsigned w0 = (1u << 16);                             // data_size=1 (2 bytes)
    if (pad_en) w0 |= (1u << 20) | (pad_int_enc << 22) | (pad_amt_enc << 25);
    g1[0] = (int)w0;
    g1[1] = (int)((tensor_d0 & 0xFFFFu) << 16);                               // dim0[15:0] @63:48
    g1[2] = (int)(((tensor_d0 >> 16) & 0xFFFFu) | ((tensor_d1 & 0xFFFFu) << 16));
    g1[3] = (int)(((tensor_d1 >> 16) & 0xFFFFu) | ((tile_d0 & 0xFFFFu) << 16));
    g1[4] = (int)(tile_d1 & 0xFFFFu);                     // tile_dim1; tile_dim2=0
    g1[5] = (int)d0_stride;                               // dim0_stride[31:0]
    g1[6] = 0;
    g1[7] = 0;
    i32x4 z4 = (i32x4){0, 0, 0, 0};
#if __clang_major__ >= 23
    i32x8 z8 = (i32x8){0, 0, 0, 0, 0, 0, 0, 0};
    __builtin_amdgcn_tensor_load_to_lds(g0, g1, z4, z4, z8, 0);
#else
    __builtin_amdgcn_tensor_load_to_lds(g0, g1, z4, z4, 0);
#endif
}
#endif

// ---------------- fp32 -> bf16 cast ----------------
__global__ __launch_bounds__(256)
void cast_f32_bf16_kernel(const float* __restrict__ in, bf16* __restrict__ out, int n) {
    int i = blockIdx.x * blockDim.x + threadIdx.x;
    int stride = gridDim.x * blockDim.x;
    for (; i < n; i += stride) out[i] = (bf16)in[i];
}

// ---------------- GEMM: C[M,N] = A[M,K] * B[K,N] + bias ----------------
// A,B bf16 row-major; fp32 accumulate; output bf16 or fp32.
// Block = 256 threads (8 waves), tile 128x128, BK=32.
// A tile staged by TDM DMA (pad fields create the 40-half LDS pitch).
#define LDT 40  // LDS row stride (halfs): 32 data + 4-DWORD pad, 16B-aligned

template <bool OUT_F32>
__global__ __launch_bounds__(256)
void gemm_bias_kernel(const bf16* __restrict__ A, const bf16* __restrict__ B,
                      const float* __restrict__ bias,
                      bf16* __restrict__ Cb, float* __restrict__ Cf,
                      int M, int N, int K) {
    __shared__ bf16 As[128 * LDT];
    __shared__ bf16 Bs[128 * LDT];

    const int tid  = threadIdx.x;
    const int lane = tid & 31;
    const int wid  = tid >> 5;
    const int l16  = lane & 15;
    const int lh   = lane >> 4;          // lane half: 0/1
    const int kofs = lh * 8;             // fragment K sub-offset (halfs)

    const int bm = blockIdx.y * 128;
    const int bn = blockIdx.x * 128;
    const int wm = (wid >> 2) * 64;      // wave M offset: 0/64
    const int wn = (wid & 3)  * 32;      // wave N offset: 0/32/64/96

    v8f acc[4][2];
#pragma unroll
    for (int i = 0; i < 4; ++i)
#pragma unroll
        for (int j = 0; j < 2; ++j) acc[i][j] = (v8f){};

    for (int k0 = 0; k0 < K; k0 += 32) {
        // ---- A tile 128x32 -> As[row][k], pitch LDT ----
#if HAVE_TDM
        if (wid == 0) {
            // pad_int_enc=3: pad every 16 DWORDs (= 32-half row);
            // pad_amt_enc=3: skip 4 DWORDs (= 8 halfs) -> pitch 40 halfs.
            tdm_load_2d_bf16((unsigned)(uintptr_t)&As[0],
                             A + (size_t)bm * K + k0,
                             (unsigned)K, (unsigned)M,
                             32u, 128u, (unsigned)K,
                             true, 3u, 3u);
        }
#else
        {
            int r = tid >> 1;
            int c = (tid & 1) * 16;
            const bf16* g = A + (size_t)(bm + r) * K + k0 + c;
            async_load_b128(g,     &As[r * LDT + c]);
            async_load_b128(g + 8, &As[r * LDT + c + 8]);
        }
#endif
        // ---- B tile 32x128 -> Bs[n][k] (transposed so K is contiguous) ----
        {
            int kr = tid >> 3;
            int c  = (tid & 7) * 16;
            const bf16* g = B + (size_t)(k0 + kr) * N + bn + c;
            bf8v b0 = *(const bf8v*)g;
            bf8v b1 = *(const bf8v*)(g + 8);
#pragma unroll
            for (int i = 0; i < 8; ++i) Bs[(c + i) * LDT + kr]     = b0[i];
#pragma unroll
            for (int i = 0; i < 8; ++i) Bs[(c + 8 + i) * LDT + kr] = b1[i];
        }
#if HAVE_TDM
        if (wid == 0) __builtin_amdgcn_s_wait_tensorcnt(0);
#else
        wait_async();
#endif
        __syncthreads();

        v16bf af[4], bfrag[2];
#pragma unroll
        for (int i = 0; i < 4; ++i) {
            const bf16* p = &As[(wm + i * 16 + l16) * LDT];
            af[i] = load16(p + kofs, p + 16 + kofs);
        }
#pragma unroll
        for (int j = 0; j < 2; ++j) {
            const bf16* p = &Bs[(wn + j * 16 + l16) * LDT];
            bfrag[j] = load16(p + kofs, p + 16 + kofs);
        }
#pragma unroll
        for (int i = 0; i < 4; ++i)
#pragma unroll
            for (int j = 0; j < 2; ++j)
                acc[i][j] = __builtin_amdgcn_wmma_f32_16x16x32_bf16(
                    false, af[i], false, bfrag[j], (short)0, acc[i][j], false, false);
        __syncthreads();
    }

    // epilogue: bias + store (C layout: VGPR v -> row v + 8*lane_half, col = l16)
#pragma unroll
    for (int j = 0; j < 2; ++j) {
        int col = bn + wn + j * 16 + l16;
        float bv = bias[col];
#pragma unroll
        for (int i = 0; i < 4; ++i) {
            int row0 = bm + wm + i * 16 + lh * 8;
#pragma unroll
            for (int v = 0; v < 8; ++v) {
                float val = acc[i][j][v] + bv;
                size_t idx = (size_t)(row0 + v) * N + col;
                if (OUT_F32) Cf[idx] = val;
                else         Cb[idx] = (bf16)val;
            }
        }
    }
}

// ---------------- Flash attention ----------------
// grid = (B*H, S/128); block = 256 threads (8 waves), each wave owns 16 query rows.
// Q/K tiles staged with global_load_async_to_lds_b128 (ASYNCcnt path).
__global__ __launch_bounds__(256)
void attn_kernel(const bf16* __restrict__ qkv, bf16* __restrict__ out) {
    __shared__ bf16 Qs[128 * HD];        // 16 KB, row-major [q][d]
    __shared__ bf16 Ks[128 * HD];        // 16 KB, row-major [k][d]
    __shared__ bf16 Vts[HD * 128];       // 16 KB, transposed [d][k]
    __shared__ bf16 Ps[8][16 * 32];      // 8 KB, per-wave P chunk [m][k]

    const int tid  = threadIdx.x;
    const int lane = tid & 31;
    const int wid  = tid >> 5;
    const int l16  = lane & 15;
    const int lh   = lane >> 4;
    const int kofs = lh * 8;

    const int bh = blockIdx.x;           // 0..63
    const int b  = bh >> 4;
    const int h  = bh & 15;
    const int qt = blockIdx.y;           // 0..15

    const int tokQ = b * SQ + qt * 128;
    const int qcol = h * (3 * HD);
    const int kcol = qcol + HD;
    const int vcol = qcol + 2 * HD;

    // ---- load Q tile (async -> LDS) ----
    {
        int r = tid >> 1;
        int c = (tid & 1) * 32;
        const bf16* g = qkv + (size_t)(tokQ + r) * QKVC + qcol + c;
        bf16* l = &Qs[r * HD + c];
#pragma unroll
        for (int i = 0; i < 4; ++i) async_load_b128(g + i * 8, l + i * 8);
    }
    wait_async();
    __syncthreads();

    // wave's Q fragments (M = wid*16 + l16, K split into two 32-chunks)
    const int qrow = wid * 16 + l16;
    v16bf aq[2];
#pragma unroll
    for (int c = 0; c < 2; ++c) {
        const bf16* p = &Qs[qrow * HD + c * 32];
        aq[c] = load16(p + kofs, p + 16 + kofs);
    }

    float m_i[8], l_i[8];
    v8f   O[4];
#pragma unroll
    for (int v = 0; v < 8; ++v) { m_i[v] = -1e30f; l_i[v] = 0.0f; }
#pragma unroll
    for (int nd = 0; nd < 4; ++nd) O[nd] = (v8f){};

    for (int kt = 0; kt < SQ / 128; ++kt) {
        __syncthreads();
        const int tokK = b * SQ + kt * 128;
        // K tile (async -> LDS, row-major)
        {
            int r = tid >> 1;
            int c = (tid & 1) * 32;
            const bf16* g = qkv + (size_t)(tokK + r) * QKVC + kcol + c;
            bf16* l = &Ks[r * HD + c];
#pragma unroll
            for (int i = 0; i < 4; ++i) async_load_b128(g + i * 8, l + i * 8);
        }
        // V tile transposed -> Vts[d][k] (needs per-element scatter)
        {
            int r = tid >> 1;
            int c = (tid & 1) * 32;
            const bf16* g = qkv + (size_t)(tokK + r) * QKVC + vcol + c;
#pragma unroll
            for (int i = 0; i < 4; ++i) {
                bf8v vv = *(const bf8v*)(g + i * 8);
#pragma unroll
                for (int e = 0; e < 8; ++e) Vts[(c + i * 8 + e) * 128 + r] = vv[e];
            }
        }
        wait_async();
        __syncthreads();

        // ---- S = Q * K^T  (16 x 128 per wave) ----
        v8f s[8];
#pragma unroll
        for (int j = 0; j < 8; ++j) {
            s[j] = (v8f){};
#pragma unroll
            for (int c = 0; c < 2; ++c) {
                const bf16* p = &Ks[(j * 16 + l16) * HD + c * 32];
                v16bf bk = load16(p + kofs, p + 16 + kofs);
                s[j] = __builtin_amdgcn_wmma_f32_16x16x32_bf16(
                    false, aq[c], false, bk, (short)0, s[j], false, false);
            }
        }

        // ---- online softmax (rows live inside a 16-lane half) ----
        float rmax[8];
#pragma unroll
        for (int v = 0; v < 8; ++v) {
            float mv = -1e30f;
#pragma unroll
            for (int j = 0; j < 8; ++j) {
                s[j][v] *= 0.125f;               // 1/sqrt(HD)
                mv = fmaxf(mv, s[j][v]);
            }
            rmax[v] = mv;
        }
#pragma unroll
        for (int d = 1; d < 16; d <<= 1)
#pragma unroll
            for (int v = 0; v < 8; ++v)
                rmax[v] = fmaxf(rmax[v], __shfl_xor(rmax[v], d, 32));

        float alpha[8], rsum[8];
#pragma unroll
        for (int v = 0; v < 8; ++v) {
            float mn = fmaxf(m_i[v], rmax[v]);
            alpha[v] = __expf(m_i[v] - mn);
            m_i[v]   = mn;
            rsum[v]  = 0.0f;
        }
#pragma unroll
        for (int j = 0; j < 8; ++j)
#pragma unroll
            for (int v = 0; v < 8; ++v) {
                float p = __expf(s[j][v] - m_i[v]);
                s[j][v] = p;
                rsum[v] += p;
            }
#pragma unroll
        for (int d = 1; d < 16; d <<= 1)
#pragma unroll
            for (int v = 0; v < 8; ++v)
                rsum[v] += __shfl_xor(rsum[v], d, 32);
#pragma unroll
        for (int v = 0; v < 8; ++v) l_i[v] = l_i[v] * alpha[v] + rsum[v];
#pragma unroll
        for (int nd = 0; nd < 4; ++nd)
#pragma unroll
            for (int v = 0; v < 8; ++v) O[nd][v] *= alpha[v];

        // ---- O += P * V, 32-key chunks via per-wave LDS round-trip ----
        bf16* pbuf = &Ps[wid][0];            // [16][32], K contiguous
#pragma unroll
        for (int c2 = 0; c2 < 4; ++c2) {
#pragma unroll
            for (int jj = 0; jj < 2; ++jj) {
                int j = c2 * 2 + jj;
#pragma unroll
                for (int v = 0; v < 8; ++v)
                    pbuf[(v + 8 * lh) * 32 + jj * 16 + l16] = (bf16)s[j][v];
            }
            // A fragment of P (same-wave LDS ops are in-order)
            const bf16* pp = &pbuf[l16 * 32];
            v16bf ap = load16(pp + kofs, pp + 16 + kofs);
#pragma unroll
            for (int nd = 0; nd < 4; ++nd) {
                const bf16* vp = &Vts[(nd * 16 + l16) * 128 + c2 * 32];
                v16bf bv = load16(vp + kofs, vp + 16 + kofs);
                O[nd] = __builtin_amdgcn_wmma_f32_16x16x32_bf16(
                    false, ap, false, bv, (short)0, O[nd], false, false);
            }
        }
    }

    // ---- normalize + store ----
    float invl[8];
#pragma unroll
    for (int v = 0; v < 8; ++v) invl[v] = 1.0f / l_i[v];
#pragma unroll
    for (int nd = 0; nd < 4; ++nd) {
        int col = h * HD + nd * 16 + l16;
#pragma unroll
        for (int v = 0; v < 8; ++v) {
            int row = tokQ + wid * 16 + v + 8 * lh;
            out[(size_t)row * DEMB + col] = (bf16)(O[nd][v] * invl[v]);
        }
    }
}

// ---------------- launch ----------------
extern "C" void kernel_launch(void* const* d_in, const int* in_sizes, int n_in,
                              void* d_out, int out_size, void* d_ws, size_t ws_size,
                              hipStream_t stream) {
    const float* x    = (const float*)d_in[0];
    const float* Wqkv = (const float*)d_in[1];
    const float* bqkv = (const float*)d_in[2];
    const float* Wo   = (const float*)d_in[3];
    const float* bo   = (const float*)d_in[4];
    float* out        = (float*)d_out;

    char* ws = (char*)d_ws;
    bf16* xb    = (bf16*)ws;                                        // 16 MB
    bf16* wqkvb = (bf16*)(ws + (size_t)NTOK * DIN * 2);             //  6 MB
    bf16* wob   = (bf16*)((char*)wqkvb + (size_t)DIN * QKVC * 2);   //  2 MB
    bf16* qkvb  = (bf16*)((char*)wob   + (size_t)DEMB * DEMB * 2);  // 48 MB
    bf16* attnb = (bf16*)((char*)qkvb  + (size_t)NTOK * QKVC * 2);  // 16 MB

    // fp32 -> bf16 casts
    cast_f32_bf16_kernel<<<4096, 256, 0, stream>>>(x,    xb,    NTOK * DIN);
    cast_f32_bf16_kernel<<<4096, 256, 0, stream>>>(Wqkv, wqkvb, DIN * QKVC);
    cast_f32_bf16_kernel<<<2048, 256, 0, stream>>>(Wo,   wob,   DEMB * DEMB);

    // QKV projection: [8192,1024] x [1024,3072] + b_qkv -> bf16
    gemm_bias_kernel<false><<<dim3(QKVC / 128, NTOK / 128), 256, 0, stream>>>(
        xb, wqkvb, bqkv, qkvb, nullptr, NTOK, QKVC, DIN);

    // flash attention per (b,h,q-tile)
    attn_kernel<<<dim3(BQ * NH, SQ / 128), 256, 0, stream>>>(qkvb, attnb);

    // output projection: [8192,1024] x [1024,1024] + b_o -> fp32
    gemm_bias_kernel<true><<<dim3(DEMB / 128, NTOK / 128), 256, 0, stream>>>(
        attnb, wob, bo, nullptr, out, NTOK, DEMB, DEMB);
}
